// MyopicAttention_62354335203628
// MI455X (gfx1250) — matrix operand order repaired and checked by reading, not compiled, not run on hardware
//
#include <hip/hip_runtime.h>
#include <hip/hip_bf16.h>

typedef __bf16 v16bf __attribute__((ext_vector_type(16)));
typedef float  v8f   __attribute__((ext_vector_type(8)));
typedef unsigned int u32x4 __attribute__((ext_vector_type(4)));
typedef int          i32x8 __attribute__((ext_vector_type(8)));
typedef int          i32x4 __attribute__((ext_vector_type(4)));
typedef int          vec4i __attribute__((vector_size(16)));

union FragU { uint4 u[2]; v16bf f; __bf16 h[16]; };

__device__ inline __bf16 to_bf16(float x) { return (__bf16)x; }

#if __has_builtin(__builtin_amdgcn_tensor_load_to_lds)
#define HAVE_TDM 1
#else
#define HAVE_TDM 0
#endif

#if __has_builtin(__builtin_amdgcn_global_load_async_to_lds_b128)
#define HAVE_ASYNC 1
#else
#define HAVE_ASYNC 0
#endif

// async-copy builtin wants pointers to 16B int vectors in explicit addrspaces
#define AS1V4(p) ((__attribute__((address_space(1))) vec4i*)(p))
#define AS3V4(p) ((__attribute__((address_space(3))) vec4i*)(p))

__device__ inline void wait_asynccnt0() {
#if __has_builtin(__builtin_amdgcn_s_wait_asynccnt)
    __builtin_amdgcn_s_wait_asynccnt(0);
#else
    asm volatile("s_wait_asynccnt 0x0" ::: "memory");
#endif
}

// ---------------------------------------------------------------------------
// Kernel 0: f32 -> bf16 convert
// ---------------------------------------------------------------------------
__global__ __launch_bounds__(256) void cvt_f32_bf16(const float* __restrict__ in,
                                                    __bf16* __restrict__ out, int n) {
    int i = blockIdx.x * 256 + threadIdx.x;
    if (i < n) out[i] = to_bf16(in[i]);
}

// ---------------------------------------------------------------------------
// Kernel 1/4: WMMA GEMM, C[M,Ncols] = A[M,K](bf16) * B[K,Ncols](bf16) + bias
// mode 0: store f32 to outF                       (final @Wo + bo)
// mode 1: scatter bf16 to q/k/v [B,H,N,D]         (QKV projection)
// Block: 128 thr = 4 waves; 64x64 C tile; 32-wide k-steps.
// A tile staged by the Tensor Data Mover (padded rows, stride 40 elems).
// ---------------------------------------------------------------------------
__global__ __launch_bounds__(128)
void gemm_wmma(const __bf16* __restrict__ A, const __bf16* __restrict__ Bm,
               const float* __restrict__ bias, int M, int Ncols, int K,
               int mode, float* __restrict__ outF,
               __bf16* __restrict__ qo, __bf16* __restrict__ ko, __bf16* __restrict__ vo)
{
    __shared__ __bf16 Asm[64 * 40];   // [row][k]  stride 40 (80B, 16B aligned)
    __shared__ __bf16 Bsm[64 * 40];   // [col][k]  transposed

    const int tid  = threadIdx.x;
    const int lane = tid & 31;
    const int wave = tid >> 5;
    const int wr = wave >> 1, wc = wave & 1;
    const int rowBase = blockIdx.y * 64;
    const int colBase = blockIdx.x * 64;
    const int m16 = lane & 15;
    const int half = lane >> 4;
    (void)M; (void)lane;

    v8f acc[2][2] = {};

    for (int k0 = 0; k0 < K; k0 += 32) {
        // ---- Stage B tile 32x64 transposed -> Bsm[col][k] ----
        {
            int kr0 = tid >> 3, seg = tid & 7;   // chunk c0 = tid
            int kr1 = kr0 + 16;                  // chunk c1 = tid + 128
            union { uint4 u; __bf16 h[8]; } t0, t1;
            t0.u = *(const uint4*)(Bm + (size_t)(k0 + kr0) * Ncols + colBase + seg * 8);
            t1.u = *(const uint4*)(Bm + (size_t)(k0 + kr1) * Ncols + colBase + seg * 8);
#pragma unroll
            for (int e = 0; e < 8; ++e) Bsm[(seg * 8 + e) * 40 + kr0] = t0.h[e];
#pragma unroll
            for (int e = 0; e < 8; ++e) Bsm[(seg * 8 + e) * 40 + kr1] = t1.h[e];
        }

        // ---- Stage A tile 64x32 -> Asm[row][k] ----
#if HAVE_TDM
        if (wave == 0) {
            // TDM D#: 2-D bf16 tile 32x64, row stride K elems in memory,
            // LDS rows padded 16DW data + 4DW pad = 40-element stride.
            unsigned long long ga =
                (unsigned long long)(const void*)(A + (size_t)rowBase * K + k0);
            unsigned ldsA = (unsigned)(unsigned long long)(const void*)&Asm[0];
            u32x4 g0;
            g0.x = 1u;                                       // count=1, load mode
            g0.y = ldsA;                                     // lds_addr
            g0.z = (unsigned)(ga & 0xFFFFFFFFull);           // global_addr[31:0]
            g0.w = (unsigned)((ga >> 32) & 0x01FFFFFFull)    // global_addr[56:32]
                   | (2u << 30);                             // type = 2 (image)
            i32x8 g1;
            g1[0] = (int)((1u << 16)      // data_size = 2 bytes
                        | (1u << 20)      // pad_enable
                        | (3u << 22)      // pad_interval: 16 DWORDs
                        | (3u << 25));    // pad_amount:    4 DWORDs
            g1[1] = (int)(32u << 16);     // tensor_dim0 = 32 (low16)
            g1[2] = (int)(64u << 16);     // tensor_dim1 = 64 (low16)
            g1[3] = (int)(32u << 16);     // tile_dim0 = 32
            g1[4] = 64;                   // tile_dim1 = 64, tile_dim2 = 0 (2-D)
            g1[5] = K;                    // tensor_dim0_stride[31:0] = K elems
            g1[6] = 0;
            g1[7] = 0;
            i32x4 z4 = {0, 0, 0, 0};
            i32x8 z8 = {0, 0, 0, 0, 0, 0, 0, 0};
            __builtin_amdgcn_tensor_load_to_lds(g0, g1, z4, z4, z8, 0);
            __builtin_amdgcn_s_wait_tensorcnt(0);
        }
#else
        {
            int r0 = tid >> 2, seg = tid & 3;    // chunk c0 = tid
            int r1 = r0 + 32;                    // chunk c1 = tid + 128
            uint4 a0 = *(const uint4*)(A + (size_t)(rowBase + r0) * K + k0 + seg * 8);
            uint4 a1 = *(const uint4*)(A + (size_t)(rowBase + r1) * K + k0 + seg * 8);
            *(uint4*)(&Asm[r0 * 40 + seg * 8]) = a0;
            *(uint4*)(&Asm[r1 * 40 + seg * 8]) = a1;
        }
#endif
        if (k0 + 32 < K) {  // global_prefetch_b8 of next tiles
            __builtin_prefetch(A + (size_t)(rowBase + (tid >> 1)) * K + k0 + 32, 0, 1);
            __builtin_prefetch(Bm + (size_t)(k0 + 32 + (tid & 31)) * Ncols + colBase, 0, 1);
        }
        __syncthreads();

        FragU a[2], b[2];
#pragma unroll
        for (int i = 0; i < 2; ++i) {
            const __bf16* p = &Asm[(wr * 32 + i * 16 + m16) * 40];
            a[i].u[0] = *(const uint4*)(p + half * 8);
            a[i].u[1] = *(const uint4*)(p + 16 + half * 8);
        }
#pragma unroll
        for (int j = 0; j < 2; ++j) {
            const __bf16* p = &Bsm[(wc * 32 + j * 16 + m16) * 40];
            b[j].u[0] = *(const uint4*)(p + half * 8);
            b[j].u[1] = *(const uint4*)(p + 16 + half * 8);
        }
#pragma unroll
        for (int i = 0; i < 2; ++i)
#pragma unroll
            for (int j = 0; j < 2; ++j)
                acc[i][j] = __builtin_amdgcn_wmma_f32_16x16x32_bf16(
                    false, a[i].f, false, b[j].f, (short)0, acc[i][j], false, false);
        __syncthreads();
    }

    // Epilogue
#pragma unroll
    for (int i = 0; i < 2; ++i)
#pragma unroll
        for (int j = 0; j < 2; ++j)
#pragma unroll
            for (int r = 0; r < 8; ++r) {
                int row = rowBase + wr * 32 + i * 16 + r + 8 * half;
                int col = colBase + wc * 32 + j * 16 + m16;
                float val = acc[i][j][r] + bias[col];
                if (mode == 0) {
                    outF[(size_t)row * Ncols + col] = val;
                } else {
                    // col = (h*64 + d)*3 + s   (qkv innermost)
                    int s = col % 3, hd = col / 3, h = hd >> 6, d = hd & 63;
                    int bb = row >> 11, n = row & 2047;   // N = 2048
                    __bf16* dst = (s == 0) ? qo : (s == 1) ? ko : vo;
                    dst[((((size_t)bb * 8 + h) * 2048) + n) * 64 + d] = to_bf16(val);
                }
            }
}

// ---------------------------------------------------------------------------
// Kernel 2: per-(b,h,nw) radix-select of 64 smallest noised distances.
// Threshold + index-ordered compaction == top_k tie-break + sorted idx.
// ---------------------------------------------------------------------------
__global__ __launch_bounds__(256)
void select_topk(const float* __restrict__ pareto, const unsigned char* __restrict__ mask,
                 int* __restrict__ idx_out)
{
    __shared__ unsigned int ukey[2048];
    __shared__ unsigned int hist[256];
    __shared__ unsigned int scan1[256], scan2[256];
    __shared__ int s_bin, s_want;
    __shared__ unsigned int s_less;

    const int row = blockIdx.x;          // ((b*8+h)*512 + nw)
    const int nw  = row & 511;
    const int b   = row >> 12;
    const int t   = threadIdx.x;

    const float* prow = pareto + (size_t)row * 2048;
    const unsigned char* mrow = mask + (size_t)b * 2048;

    for (int j = t; j < 2048; j += 256) {
        float g = fabsf((float)(nw - (j >> 2)));
        float val = g - prow[j];
        if (mrow[j]) val = 3.402823466e38f;
        unsigned int fb = __float_as_uint(val);
        ukey[j] = (fb & 0x80000000u) ? ~fb : (fb | 0x80000000u);
    }
    __syncthreads();

    unsigned int prefix = 0;
    int want = 63;
    for (int shift = 24; shift >= 0; shift -= 8) {
        hist[t] = 0;
        __syncthreads();
        unsigned int pm = (shift == 24) ? 0u : (0xFFFFFFFFu << (shift + 8));
        for (int j = t; j < 2048; j += 256) {
            unsigned int key = ukey[j];
            if ((key & pm) == prefix) atomicAdd(&hist[(key >> shift) & 255u], 1u);
        }
        __syncthreads();
        if (t == 0) {
            unsigned int cum = 0; int bin = 255;
            for (int i2 = 0; i2 < 256; ++i2) {
                unsigned int hc = hist[i2];
                if ((unsigned int)want < cum + hc) { bin = i2; break; }
                cum += hc;
            }
            s_bin = bin; s_want = want - (int)cum;
        }
        __syncthreads();
        prefix |= ((unsigned int)s_bin) << shift;
        want = s_want;
        __syncthreads();
    }
    const unsigned int T = prefix;

    if (t == 0) s_less = 0;
    __syncthreads();
    unsigned int lcl = 0;
    for (int j = t; j < 2048; j += 256) lcl += (ukey[j] < T);
    if (lcl) atomicAdd(&s_less, lcl);
    __syncthreads();
    const unsigned int tieBudget = 64u - s_less;

    const int base = t * 8;
    unsigned int cs = 0, ctn = 0;
    for (int j = base; j < base + 8; ++j) {
        unsigned int key = ukey[j];
        cs += (key < T); ctn += (key == T);
    }
    scan1[t] = cs; scan2[t] = ctn;
    __syncthreads();
    for (int off = 1; off < 256; off <<= 1) {
        unsigned int a1 = 0, a2 = 0;
        if (t >= off) { a1 = scan1[t - off]; a2 = scan2[t - off]; }
        __syncthreads();
        scan1[t] += a1; scan2[t] += a2;
        __syncthreads();
    }
    unsigned int sB = scan1[t] - cs;
    unsigned int tB = scan2[t] - ctn;
    int* orow = idx_out + (size_t)row * 64;
    for (int j = base; j < base + 8; ++j) {
        unsigned int key = ukey[j];
        if (key < T) {
            unsigned int tkept = (tB < tieBudget) ? tB : tieBudget;
            orow[sB + tkept] = j; sB++;
        } else if (key == T) {
            if (tB < tieBudget) orow[sB + tB] = j;
            tB++;
        }
    }
}

// ---------------------------------------------------------------------------
// Kernel 3: windowed attention. 1 wave per (b,h,nw) window, 4 waves/block.
// V gather goes through GLOBAL_LOAD_ASYNC_TO_LDS_B128 (ASYNCcnt path).
// ---------------------------------------------------------------------------
__global__ __launch_bounds__(128)
void attn_kernel(const __bf16* __restrict__ q, const __bf16* __restrict__ k,
                 const __bf16* __restrict__ v, const int* __restrict__ idx,
                 const float* __restrict__ pos_bias, const unsigned char* __restrict__ mask,
                 __bf16* __restrict__ aob)
{
    __shared__ __bf16 smem[4 * 6912];
    const int tid  = threadIdx.x;
    const int lane = tid & 31;
    const int wave = tid >> 5;
    __bf16* q_lds  = smem + wave * 6912;   // [16][72]
    __bf16* v_lds  = q_lds + 16 * 72;      // [64][72]
    __bf16* at_lds = v_lds + 64 * 72;      // [16][72]

    const int win = blockIdx.x * 4 + wave;     // ((b*8+h)*512 + nw)
    const int nw  = win & 511;
    const int bh  = win >> 9;
    const int h   = bh & 7;
    const int b   = bh >> 3;
    const int m16 = lane & 15, half = lane >> 4;

    const __bf16* qbase = q + ((((size_t)b * 8 + h) * 2048) + nw * 4) * 64;
    const __bf16* kbase = k + (((size_t)b * 8 + h) * 2048) * 64;
    const __bf16* vbase = v + (((size_t)b * 8 + h) * 2048) * 64;
    const int* idxw = idx + (size_t)win * 64;
    const unsigned char* mrow = mask + (size_t)b * 2048;

    // Stage Q (rows 0..3 real, 4..15 zero)
    {
        uint4 val[4];
#pragma unroll
        for (int it = 0; it < 4; ++it) {
            int c = lane + it * 32;
            int r = c >> 3, seg = c & 7;
            val[it] = make_uint4(0, 0, 0, 0);
            if (r < 4) val[it] = *(const uint4*)(qbase + r * 64 + seg * 8);
        }
#pragma unroll
        for (int it = 0; it < 4; ++it) {
            int c = lane + it * 32;
            int r = c >> 3, seg = c & 7;
            *(uint4*)(&q_lds[r * 72 + seg * 8]) = val[it];
        }
    }
    // Gather Vsel rows [64][64]
#if HAVE_ASYNC
#pragma unroll
    for (int it = 0; it < 2; ++it) {
        int kk = lane + it * 32;
        const __bf16* src = vbase + (size_t)idxw[kk] * 64;
#pragma unroll
        for (int seg = 0; seg < 8; ++seg)
            __builtin_amdgcn_global_load_async_to_lds_b128(
                AS1V4(src + seg * 8), AS3V4(&v_lds[kk * 72 + seg * 8]), 0, 0);
    }
    wait_asynccnt0();
#else
#pragma unroll
    for (int it = 0; it < 2; ++it) {
        int kk = lane + it * 32;
        const __bf16* src = vbase + (size_t)idxw[kk] * 64;
        uint4 tmp[8];
#pragma unroll
        for (int seg = 0; seg < 8; ++seg) tmp[seg] = *(const uint4*)(src + seg * 8);
#pragma unroll
        for (int seg = 0; seg < 8; ++seg) *(uint4*)(&v_lds[kk * 72 + seg * 8]) = tmp[seg];
    }
#endif
    __syncthreads();

    // Q*K^T : acc[t] = 16x16 tile over keys 16t..16t+15
    v8f acc[4] = {};
    FragU a[2];
#pragma unroll
    for (int c = 0; c < 2; ++c) {
        const __bf16* p = &q_lds[m16 * 72 + 32 * c];
        a[c].u[0] = *(const uint4*)(p + half * 8);
        a[c].u[1] = *(const uint4*)(p + 16 + half * 8);
    }
#pragma unroll
    for (int t = 0; t < 4; ++t) {
        const __bf16* krow = kbase + (size_t)idxw[16 * t + m16] * 64;
#pragma unroll
        for (int c = 0; c < 2; ++c) {
            FragU bf;
            const __bf16* p = krow + 32 * c;
            bf.u[0] = *(const uint4*)(p + half * 8);
            bf.u[1] = *(const uint4*)(p + 16 + half * 8);
            acc[t] = __builtin_amdgcn_wmma_f32_16x16x32_bf16(
                false, a[c].f, false, bf.f, (short)0, acc[t], false, false);
        }
    }

    // scale + pos_bias + mask (valid rows m<4 only; pad rows -> 0)
    const float scale = 0.125f;   // 64^-0.5
    const float BIG = 3.402823466e38f;
#pragma unroll
    for (int t = 0; t < 4; ++t) {
        int key = idxw[16 * t + m16];
        bool km = mrow[key] != 0;
#pragma unroll
        for (int r = 0; r < 8; ++r) {
            int m = r + 8 * half;
            if (m < 4) {
                int qpos = nw * 4 + m;
                float pb = pos_bias[((size_t)h * 2048 + qpos) * 2048 + key];
                float val = acc[t][r] * scale + pb;
                if ((mrow[qpos] != 0) && km) val = -BIG;
                acc[t][r] = val;
            } else {
                acc[t][r] = 0.0f;
            }
        }
    }

    // softmax across 64 keys per row (16-lane groups per half)
#pragma unroll
    for (int r = 0; r < 8; ++r) {
        float mx = fmaxf(fmaxf(acc[0][r], acc[1][r]), fmaxf(acc[2][r], acc[3][r]));
        for (int off = 8; off >= 1; off >>= 1) mx = fmaxf(mx, __shfl_xor(mx, off, 16));
        float s = 0.f;
#pragma unroll
        for (int t = 0; t < 4; ++t) { float e = __expf(acc[t][r] - mx); acc[t][r] = e; s += e; }
        for (int off = 8; off >= 1; off >>= 1) s += __shfl_xor(s, off, 16);
        float inv = 1.0f / s;
#pragma unroll
        for (int t = 0; t < 4; ++t) acc[t][r] *= inv;
    }

    // restage attn as A operand (pad rows zeroed)
#pragma unroll
    for (int t = 0; t < 4; ++t)
#pragma unroll
        for (int r = 0; r < 8; ++r) {
            int m = r + 8 * half;
            at_lds[m * 72 + 16 * t + m16] = to_bf16((m < 4) ? acc[t][r] : 0.0f);
        }
    __syncthreads();

    // out = attn(16x64) x Vsel(64x64)
    v8f oacc[4] = {};
    FragU aa[2];
#pragma unroll
    for (int c = 0; c < 2; ++c) {
        const __bf16* p = &at_lds[m16 * 72 + 32 * c];
        aa[c].u[0] = *(const uint4*)(p + half * 8);
        aa[c].u[1] = *(const uint4*)(p + 16 + half * 8);
    }
#pragma unroll
    for (int j = 0; j < 4; ++j) {
        int dim = 16 * j + m16;
#pragma unroll
        for (int c = 0; c < 2; ++c) {
            FragU bf;
#pragma unroll
            for (int vv = 0; vv < 8; ++vv) {
                int kk = 32 * c + half * 8 + 2 * vv + (vv >= 4 ? 8 : 0);
                bf.h[2 * vv]     = v_lds[kk * 72 + dim];
                bf.h[2 * vv + 1] = v_lds[(kk + 1) * 72 + dim];
            }
            oacc[j] = __builtin_amdgcn_wmma_f32_16x16x32_bf16(
                false, aa[c].f, false, bf.f, (short)0, oacc[j], false, false);
        }
    }
#pragma unroll
    for (int j = 0; j < 4; ++j)
#pragma unroll
        for (int r = 0; r < 8; ++r) {
            int m = r + 8 * half;
            if (m < 4) {
                size_t row = (size_t)b * 2048 + nw * 4 + m;
                aob[row * 512 + h * 64 + 16 * j + m16] = to_bf16(oacc[j][r]);
            }
        }
}

// ---------------------------------------------------------------------------
extern "C" void kernel_launch(void* const* d_in, const int* in_sizes, int n_in,
                              void* d_out, int out_size, void* d_ws, size_t ws_size,
                              hipStream_t stream) {
    const float* x            = (const float*)d_in[0];
    const unsigned char* mask = (const unsigned char*)d_in[1];   // jax bool = 1 byte
    const float* pos_bias     = (const float*)d_in[2];
    const float* pareto       = (const float*)d_in[3];
    const float* Wqkv         = (const float*)d_in[4];
    const float* bqkv         = (const float*)d_in[5];
    const float* Wo           = (const float*)d_in[6];
    const float* bo           = (const float*)d_in[7];
    float* out = (float*)d_out;
    char* ws = (char*)d_ws;

    __bf16* xb    = (__bf16*)(ws + 0);          // 8,388,608 B  (x bf16)
    __bf16* wqkvb = (__bf16*)(ws + 8388608);    // 1,572,864 B
    __bf16* wob   = (__bf16*)(ws + 9961472);    //   524,288 B
    __bf16* qb    = (__bf16*)(ws + 10485760);   // 8,388,608 B  [B,H,N,D]
    __bf16* kb    = (__bf16*)(ws + 18874368);   // 8,388,608 B
    __bf16* vb    = (__bf16*)(ws + 27262976);   // 8,388,608 B
    int*    idxb  = (int*)   (ws + 35651584);   // 4,194,304 B  [B,H,NW,64]
    __bf16* aob   = (__bf16*)(ws + 39845888);   // 8,388,608 B  [B*N, 512]

    cvt_f32_bf16<<<(4194304 + 255) / 256, 256, 0, stream>>>(x, xb, 4194304);
    cvt_f32_bf16<<<(786432 + 255) / 256, 256, 0, stream>>>(Wqkv, wqkvb, 786432);
    cvt_f32_bf16<<<(262144 + 255) / 256, 256, 0, stream>>>(Wo, wob, 262144);

    dim3 g1(1536 / 64, 8192 / 64);
    gemm_wmma<<<g1, 128, 0, stream>>>(xb, wqkvb, bqkv, 8192, 1536, 512,
                                      /*mode=*/1, nullptr, qb, kb, vb);

    select_topk<<<16384, 256, 0, stream>>>(pareto, mask, idxb);

    attn_kernel<<<4096, 128, 0, stream>>>(qb, kb, vb, idxb, pos_bias, mask, aob);

    dim3 g2(512 / 64, 8192 / 64);
    gemm_wmma<<<g2, 128, 0, stream>>>(aob, wob, bo, 8192, 512, 512,
                                      /*mode=*/0, out, nullptr, nullptr, nullptr);

    (void)in_sizes; (void)n_in; (void)out_size; (void)ws_size;
}
